// MultiHeadAttention_28741921145261
// MI455X (gfx1250) — compile-verified, matrix operand order checked by reading, not run on hardware
//
#include <hip/hip_runtime.h>
#include <hip/hip_bf16.h>
#include <math.h>

typedef __attribute__((ext_vector_type(16))) _Float16 v16h;
typedef __attribute__((ext_vector_type(8)))  _Float16 v8h;
typedef __attribute__((ext_vector_type(8)))  float    v8f;
typedef __attribute__((ext_vector_type(4)))  float    v4f;
typedef __attribute__((ext_vector_type(4)))  int      v4i;

#define B_   2
#define S_   2048
#define F_   2048
#define D_   1024
#define H_   16
#define HD_  64
#define M_   4096           // B_*S_ == B_*F_
#define EPS_ 1e-5f

#define AS1 __attribute__((address_space(1)))
#define AS3 __attribute__((address_space(3)))

// ---- CDNA5 async copy: 16B global -> LDS, tracked by ASYNCcnt ------------
static __device__ __forceinline__ void async_cp16(const _Float16* g, _Float16* l) {
#if __has_builtin(__builtin_amdgcn_global_load_async_to_lds_b128)
  __builtin_amdgcn_global_load_async_to_lds_b128(
      (AS1 v4i*)g, (AS3 v4i*)l, 0, 0);
#else
  *(v8h*)l = *(const v8h*)g;
#endif
}
template <int N>
static __device__ __forceinline__ void async_wait() {
#if __has_builtin(__builtin_amdgcn_s_wait_asynccnt)
  __builtin_amdgcn_s_wait_asynccnt(N);
#endif
}

// Concatenate two contiguous 8-half K-runs into a WMMA 16-bit fragment.
static __device__ __forceinline__ v16h frag_from(const _Float16* row, int half16) {
  v8h lo = *(const v8h*)(row + 8 * half16);        // K = {0..7}  + 8*half
  v8h hi = *(const v8h*)(row + 16 + 8 * half16);   // K = {16..23}+ 8*half
  return __builtin_shufflevector(lo, hi, 0, 1, 2, 3, 4, 5, 6, 7,
                                 8, 9, 10, 11, 12, 13, 14, 15);
}

// ---------------------------------------------------------------------------
// Elementwise f32 -> f16 (8 elems/thread).
// ---------------------------------------------------------------------------
__global__ __launch_bounds__(256) void cvt_f16(
    const float* __restrict__ in, _Float16* __restrict__ out) {
  size_t i = ((size_t)blockIdx.x * 256 + threadIdx.x) * 8;
  v4f a0 = *(const v4f*)(in + i);
  v4f a1 = *(const v4f*)(in + i + 4);
  v8h h;
#pragma unroll
  for (int e = 0; e < 4; ++e) { h[e] = (_Float16)a0[e]; h[4 + e] = (_Float16)a1[e]; }
  *(v8h*)(out + i) = h;
}

// ---------------------------------------------------------------------------
// Transpose + convert: in (R x C) f32 -> out (C x R) f16, batched over z.
// ---------------------------------------------------------------------------
__global__ __launch_bounds__(256) void transpose_cvt(
    const float* __restrict__ in, _Float16* __restrict__ out, int R, int C) {
  __shared__ float t[32][33];
  const int rb = blockIdx.y * 32, cb = blockIdx.x * 32;
  const float*   inb  = in  + (size_t)blockIdx.z * R * C;
  _Float16*      outb = out + (size_t)blockIdx.z * R * C;
  const int x = threadIdx.x, y = threadIdx.y;   // (32, 8)
#pragma unroll
  for (int j = 0; j < 4; ++j)
    t[y + j * 8][x] = inb[(size_t)(rb + y + j * 8) * C + cb + x];
  __syncthreads();
#pragma unroll
  for (int j = 0; j < 4; ++j)
    outb[(size_t)(cb + y + j * 8) * R + rb + x] = (_Float16)t[x][y + j * 8];
}

// ---------------------------------------------------------------------------
// C[4096,1024](f32) = A[4096,1024](f16) @ Wt[1024,1024](f16, N-major rows)
// Workgroup tile 64(M) x 128(N), 8 waves of 32x32, K stepped by 32.
// Double-buffered async b128 staging (copies overlap WMMA).
// ---------------------------------------------------------------------------
__global__ __launch_bounds__(256) void gemm_f16wmma(
    const _Float16* __restrict__ A, const _Float16* __restrict__ Wt,
    float* __restrict__ C) {
  __shared__ _Float16 As[2][64][40];     // 64 M-rows x 32 K (80B rows)
  __shared__ _Float16 Bts[2][128][40];   // 128 N-rows x 32 K
  const int tid  = threadIdx.x;
  const int lane = tid & 31;
  const int wave = tid >> 5;
  const int m0 = blockIdx.y * 64;
  const int n0 = blockIdx.x * 128;
  const int wm = (wave & 1) * 32;
  const int wn = (wave >> 1) * 32;
  const int half16 = lane >> 4;
  const int arow   = lane & 15;

  // per-thread staging segments (3 x 16B async per thread per K slab)
  const int ar = tid >> 2, ac = (tid & 3) * 8;
  const int br0 = tid >> 2,         bc0 = (tid & 3) * 8;
  const int br1 = (tid + 256) >> 2, bc1 = (tid & 3) * 8;

  // prologue: stage slab 0 into buffer 0
  async_cp16(&A[(size_t)(m0 + ar) * D_ + ac],    &As[0][ar][ac]);
  async_cp16(&Wt[(size_t)(n0 + br0) * D_ + bc0], &Bts[0][br0][bc0]);
  async_cp16(&Wt[(size_t)(n0 + br1) * D_ + bc1], &Bts[0][br1][bc1]);

  v8f acc[2][2] = {};

  for (int k0 = 0, it = 0; k0 < D_; k0 += 32, ++it) {
    const int cur = it & 1;
    if (k0 + 32 < D_) {
      const int nxt = cur ^ 1, kn = k0 + 32;
      async_cp16(&A[(size_t)(m0 + ar) * D_ + kn + ac],    &As[nxt][ar][ac]);
      async_cp16(&Wt[(size_t)(n0 + br0) * D_ + kn + bc0], &Bts[nxt][br0][bc0]);
      async_cp16(&Wt[(size_t)(n0 + br1) * D_ + kn + bc1], &Bts[nxt][br1][bc1]);
      if (k0 + 64 < D_) {
        __builtin_prefetch(&A[(size_t)(m0 + ar) * D_ + k0 + 64 + ac], 0, 1);
        __builtin_prefetch(&Wt[(size_t)(n0 + br0) * D_ + k0 + 64 + bc0], 0, 1);
      }
      async_wait<3>();    // current slab done; next slab still in flight
    } else {
      async_wait<0>();
    }
    __syncthreads();

    v16h afrag[2], bfrag[2];
#pragma unroll
    for (int mi = 0; mi < 2; ++mi)
      afrag[mi] = frag_from(&As[cur][wm + mi * 16 + arow][0], half16);
#pragma unroll
    for (int ni = 0; ni < 2; ++ni)
      bfrag[ni] = frag_from(&Bts[cur][wn + ni * 16 + arow][0], half16);

#pragma unroll
    for (int mi = 0; mi < 2; ++mi)
#pragma unroll
      for (int ni = 0; ni < 2; ++ni)
        acc[mi][ni] = __builtin_amdgcn_wmma_f32_16x16x32_f16(
            false, afrag[mi], false, bfrag[ni], (short)0, acc[mi][ni],
            false, false);
    __syncthreads();
  }

  float* cb = &C[(size_t)(m0 + wm + half16 * 8) * D_ + n0 + wn + arow];
#pragma unroll
  for (int mi = 0; mi < 2; ++mi)
#pragma unroll
    for (int ni = 0; ni < 2; ++ni)
#pragma unroll
      for (int r = 0; r < 8; ++r)
        cb[(size_t)(mi * 16 + r) * D_ + ni * 16] = acc[mi][ni][r];
}

// ---------------------------------------------------------------------------
// Per-head LayerNorm over HD=64: f32 in -> f16 out (optionally scaled).
// One wave per head vector.
// ---------------------------------------------------------------------------
__global__ __launch_bounds__(256) void head_ln(
    const float* __restrict__ in, _Float16* __restrict__ out,
    const float* __restrict__ gamma, const float* __restrict__ beta,
    float scale) {
  const int lane = threadIdx.x & 31;
  const int g = blockIdx.x * 8 + (threadIdx.x >> 5);
  const size_t base = (size_t)g * 64;
  float x0 = in[base + lane];
  float x1 = in[base + 32 + lane];
  float s  = x0 + x1;
  float s2 = x0 * x0 + x1 * x1;
#pragma unroll
  for (int m = 16; m >= 1; m >>= 1) {
    s  += __shfl_xor(s,  m);
    s2 += __shfl_xor(s2, m);
  }
  float mean = s * (1.0f / 64.0f);
  float var  = s2 * (1.0f / 64.0f) - mean * mean;
  float rstd = rsqrtf(var + EPS_);
  out[base + lane]      = (_Float16)(((x0 - mean) * rstd * gamma[lane]      + beta[lane])      * scale);
  out[base + 32 + lane] = (_Float16)(((x1 - mean) * rstd * gamma[32 + lane] + beta[32 + lane]) * scale);
}

// ---------------------------------------------------------------------------
// Flash attention per (b,h). grid = (S/128, B*H). Each wave owns 16 Q rows.
// q,k: f16 (B,S/F,D) row-major; v: f16 d-major [(b*D+d)][F]. Output f32.
// Double-buffered async staging of K/V chunks.
// ---------------------------------------------------------------------------
__global__ __launch_bounds__(256) void flash_attn(
    const _Float16* __restrict__ q, const _Float16* __restrict__ k,
    const _Float16* __restrict__ v, float* __restrict__ o) {
  __shared__ _Float16 Qs[128][72];         // 128 s-rows x 64 d (144B rows)
  __shared__ _Float16 Ks[2][32][72];       // 32 f-rows x 64 d
  __shared__ _Float16 Vs[2][64][40];       // 64 d-rows x 32 f
  __shared__ _Float16 Ps[8][16][40];       // per-wave P relayout scratch
  const int tid  = threadIdx.x;
  const int lane = tid & 31;
  const int wave = tid >> 5;
  const int half16 = lane >> 4;
  const int arow   = lane & 15;
  const int bh = blockIdx.y;
  const int b  = bh / H_;
  const int hd = bh % H_;
  const size_t qbase = (size_t)b * S_ * D_ + (size_t)hd * HD_;
  const size_t kbase = (size_t)b * F_ * D_ + (size_t)hd * HD_;
  const size_t vbase = ((size_t)b * D_ + (size_t)hd * HD_) * F_;

  // staging segments: Q 4x16B once; K,V 1x16B each per chunk
  const int kr = tid >> 3, kc8 = (tid & 7) * 8;
  const int vd = tid >> 2, vf  = (tid & 3) * 8;

  // ---- stage Q tile (128 x 64 f16) + first K/V chunk
#pragma unroll
  for (int i = 0; i < 4; ++i) {
    int idx = tid + i * 256;
    int r = idx >> 3, c = (idx & 7) * 8;
    async_cp16(&q[qbase + (size_t)(blockIdx.x * 128 + r) * D_ + c], &Qs[r][c]);
  }
  async_cp16(&k[kbase + (size_t)kr * D_ + kc8], &Ks[0][kr][kc8]);
  async_cp16(&v[vbase + (size_t)vd * F_ + vf],  &Vs[0][vd][vf]);
  async_wait<2>();   // in-order completion: Q done, K/V chunk 0 may be in flight
  __syncthreads();

  v16h qa[2];
#pragma unroll
  for (int kc = 0; kc < 2; ++kc)
    qa[kc] = frag_from(&Qs[wave * 16 + arow][kc * 32], half16);

  v8f oacc[4] = {};
  float mrow[8], lrow[8];
#pragma unroll
  for (int r = 0; r < 8; ++r) { mrow[r] = -INFINITY; lrow[r] = 0.0f; }

  for (int f0 = 0, it = 0; f0 < F_; f0 += 32, ++it) {
    const int cur = it & 1;
    if (f0 + 32 < F_) {
      const int nxt = cur ^ 1, fn = f0 + 32;
      async_cp16(&k[kbase + (size_t)(fn + kr) * D_ + kc8], &Ks[nxt][kr][kc8]);
      async_cp16(&v[vbase + (size_t)vd * F_ + fn + vf],    &Vs[nxt][vd][vf]);
      if (f0 + 64 < F_) {
        __builtin_prefetch(&k[kbase + (size_t)(f0 + 64 + kr) * D_ + kc8], 0, 1);
        __builtin_prefetch(&v[vbase + (size_t)vd * F_ + f0 + 64 + vf], 0, 1);
      }
      async_wait<2>();   // current K/V chunk done; next chunk in flight
    } else {
      async_wait<0>();
    }
    __syncthreads();

    // S = q @ k^T : 16x32 scores, K=64 chained.
    v8f sacc[2] = {};
#pragma unroll
    for (int kc = 0; kc < 2; ++kc) {
      v16h bf[2];
#pragma unroll
      for (int ni = 0; ni < 2; ++ni)
        bf[ni] = frag_from(&Ks[cur][ni * 16 + arow][kc * 32], half16);
#pragma unroll
      for (int ni = 0; ni < 2; ++ni)
        sacc[ni] = __builtin_amdgcn_wmma_f32_16x16x32_f16(
            false, qa[kc], false, bf[ni], (short)0, sacc[ni], false, false);
    }

    // Online softmax: each row's 16 cols live in one 16-lane half-wave.
#pragma unroll
    for (int r = 0; r < 8; ++r) {
      float mx = fmaxf(sacc[0][r], sacc[1][r]);
#pragma unroll
      for (int m = 8; m >= 1; m >>= 1) mx = fmaxf(mx, __shfl_xor(mx, m));
      float mnew = fmaxf(mrow[r], mx);
      float corr = __expf(mrow[r] - mnew);
      float p0 = __expf(sacc[0][r] - mnew);
      float p1 = __expf(sacc[1][r] - mnew);
      float rs = p0 + p1;
#pragma unroll
      for (int m = 8; m >= 1; m >>= 1) rs += __shfl_xor(rs, m);
      lrow[r] = lrow[r] * corr + rs;
      mrow[r] = mnew;
#pragma unroll
      for (int c = 0; c < 4; ++c) oacc[c][r] *= corr;
      int prow = r + half16 * 8;
      Ps[wave][prow][arow]      = (_Float16)p0;
      Ps[wave][prow][16 + arow] = (_Float16)p1;
    }

    v16h pa = frag_from(&Ps[wave][arow][0], half16);

    // O += P(16x32) @ V(32x64)
#pragma unroll
    for (int c = 0; c < 4; ++c) {
      v16h bv = frag_from(&Vs[cur][c * 16 + arow][0], half16);
      oacc[c] = __builtin_amdgcn_wmma_f32_16x16x32_f16(
          false, pa, false, bv, (short)0, oacc[c], false, false);
    }
    __syncthreads();
  }

  const int s0 = blockIdx.x * 128 + wave * 16;
  float* ob = &o[((size_t)b * S_ + s0 + half16 * 8) * D_ + hd * HD_ + arow];
#pragma unroll
  for (int r = 0; r < 8; ++r) {
    float inv = 1.0f / lrow[r];
#pragma unroll
    for (int c = 0; c < 4; ++c)
      ob[(size_t)r * D_ + c * 16] = oacc[c][r] * inv;
  }
}

// ---------------------------------------------------------------------------
// LayerNorm over D=1024 per row: f32 in -> f16 out.
// ---------------------------------------------------------------------------
__global__ __launch_bounds__(256) void row_ln(
    const float* __restrict__ in, const float* __restrict__ gamma,
    const float* __restrict__ beta, _Float16* __restrict__ out) {
  __shared__ float red[16];
  const int tid  = threadIdx.x;
  const int lane = tid & 31;
  const int wave = tid >> 5;
  const size_t base = (size_t)blockIdx.x * D_;
  float s = 0.0f, s2 = 0.0f;
#pragma unroll
  for (int i = 0; i < 4; ++i) {
    float x = in[base + tid + i * 256];
    s += x; s2 += x * x;
  }
#pragma unroll
  for (int m = 16; m >= 1; m >>= 1) {
    s  += __shfl_xor(s,  m);
    s2 += __shfl_xor(s2, m);
  }
  if (lane == 0) { red[wave] = s; red[8 + wave] = s2; }
  __syncthreads();
  if (tid == 0) {
    float ts = 0.0f, ts2 = 0.0f;
#pragma unroll
    for (int w = 0; w < 8; ++w) { ts += red[w]; ts2 += red[8 + w]; }
    red[0] = ts; red[1] = ts2;
  }
  __syncthreads();
  float mean = red[0] * (1.0f / (float)D_);
  float var  = red[1] * (1.0f / (float)D_) - mean * mean;
  float rstd = rsqrtf(var + EPS_);
#pragma unroll
  for (int i = 0; i < 4; ++i) {
    int idx = tid + i * 256;
    float x = in[base + idx];
    out[base + idx] = (_Float16)((x - mean) * rstd * gamma[idx] + beta[idx]);
  }
}

// ---------------------------------------------------------------------------
extern "C" void kernel_launch(void* const* d_in, const int* in_sizes, int n_in,
                              void* d_out, int out_size, void* d_ws, size_t ws_size,
                              hipStream_t stream) {
  const float* x_q   = (const float*)d_in[0];
  const float* x_k   = (const float*)d_in[1];
  const float* x_v   = (const float*)d_in[2];
  const float* Wq    = (const float*)d_in[3];
  const float* Wk    = (const float*)d_in[4];
  const float* Wv    = (const float*)d_in[5];
  const float* Wproj = (const float*)d_in[6];
  const float* q_g   = (const float*)d_in[7];
  const float* q_b   = (const float*)d_in[8];
  const float* k_g   = (const float*)d_in[9];
  const float* k_b   = (const float*)d_in[10];
  const float* o_g   = (const float*)d_in[11];
  const float* o_b   = (const float*)d_in[12];
  float* out = (float*)d_out;

  const size_t MB = 1024 * 1024;
  char* w = (char*)d_ws;
  _Float16* xqh = (_Float16*)(w + 0 * MB);    // 8MB each
  _Float16* xkh = (_Float16*)(w + 8 * MB);
  _Float16* xvh = (_Float16*)(w + 16 * MB);
  _Float16* wqt = (_Float16*)(w + 24 * MB);   // 2MB each (transposed f16)
  _Float16* wkt = (_Float16*)(w + 26 * MB);
  _Float16* wvt = (_Float16*)(w + 28 * MB);
  _Float16* wpt = (_Float16*)(w + 30 * MB);
  float*    qb  = (float*)(w + 32 * MB);      // 16MB each (f32 projections)
  float*    kb  = (float*)(w + 48 * MB);
  float*    vb  = (float*)(w + 64 * MB);
  _Float16* qh  = (_Float16*)(w + 80 * MB);   // 8MB each (post-LN f16)
  _Float16* kh  = (_Float16*)(w + 88 * MB);
  _Float16* vt  = (_Float16*)(w + 96 * MB);   // 8MB (d-major f16 V)
  float*    ao  = (float*)(w + 104 * MB);     // 16MB attention out f32
  _Float16* lnh = (_Float16*)(w + 0 * MB);    // reuse xqh space

  dim3 gblk(256);
  dim3 ggrid(D_ / 128, M_ / 64);
  dim3 tblk(32, 8);

  // one-time precision/layout conversion
  cvt_f16<<<(M_ * D_ / 8) / 256, 256, 0, stream>>>(x_q, xqh);
  cvt_f16<<<(M_ * D_ / 8) / 256, 256, 0, stream>>>(x_k, xkh);
  cvt_f16<<<(M_ * D_ / 8) / 256, 256, 0, stream>>>(x_v, xvh);
  transpose_cvt<<<dim3(32, 32, 1), tblk, 0, stream>>>(Wq, wqt, D_, D_);
  transpose_cvt<<<dim3(32, 32, 1), tblk, 0, stream>>>(Wk, wkt, D_, D_);
  transpose_cvt<<<dim3(32, 32, 1), tblk, 0, stream>>>(Wv, wvt, D_, D_);
  transpose_cvt<<<dim3(32, 32, 1), tblk, 0, stream>>>(Wproj, wpt, D_, D_);

  // projections
  gemm_f16wmma<<<ggrid, gblk, 0, stream>>>(xqh, wqt, qb);
  gemm_f16wmma<<<ggrid, gblk, 0, stream>>>(xkh, wkt, kb);
  gemm_f16wmma<<<ggrid, gblk, 0, stream>>>(xvh, wvt, vb);

  // per-head LN -> f16; V -> d-major f16
  head_ln<<<(B_ * S_ * H_) / 8, 256, 0, stream>>>(qb, qh, q_g, q_b, 0.125f);
  head_ln<<<(B_ * F_ * H_) / 8, 256, 0, stream>>>(kb, kh, k_g, k_b, 1.0f);
  transpose_cvt<<<dim3(D_ / 32, F_ / 32, B_), tblk, 0, stream>>>(vb, vt, F_, D_);

  flash_attn<<<dim3(S_ / 128, B_ * H_), 256, 0, stream>>>(qh, kh, vt, ao);

  row_ln<<<M_, 256, 0, stream>>>(ao, o_g, o_b, lnh);

  gemm_f16wmma<<<ggrid, gblk, 0, stream>>>(lnh, wpt, out);
}
